// MultiHeadAttentionLayer_8151847927922
// MI455X (gfx1250) — matrix operand order repaired and checked by reading, not compiled, not run on hardware
//
#include <hip/hip_runtime.h>
#include <hip/hip_bf16.h>

#define B_     4
#define S_     2048
#define HID_   1024
#define HEADS_ 16
#define D_     64

typedef __attribute__((ext_vector_type(16))) __bf16   v16bf;
typedef __attribute__((ext_vector_type(2)))  __bf16   bf16x2;
typedef __attribute__((ext_vector_type(8)))  float    v8f;
typedef __attribute__((ext_vector_type(4)))  unsigned uint4v;
typedef __attribute__((ext_vector_type(2)))  unsigned uint2v;
typedef __attribute__((ext_vector_type(4)))  float    float4v;

__device__ __forceinline__ __bf16 f2bf(float f) { return (__bf16)f; }   // native cvt
__device__ __forceinline__ unsigned pack2bf(float a, float b) {
  bf16x2 p; p.x = (__bf16)a; p.y = (__bf16)b;                           // v_cvt_pk_bf16_f32
  return __builtin_bit_cast(unsigned, p);
}

union FragU { uint4v q[2]; v16bf v; };
// two contiguous 16B-aligned 8-element bf16 runs -> one WMMA operand
__device__ __forceinline__ v16bf ld_frag(const __bf16* p0, const __bf16* p1) {
  FragU u;
  u.q[0] = *(const uint4v*)p0;
  u.q[1] = *(const uint4v*)p1;
  return u.v;
}

// async 16B global -> LDS copy (ASYNCcnt-tracked, no VGPR round trip)
__device__ __forceinline__ void async_cp16(void* lds, const void* gaddr) {
  unsigned l = (unsigned)(unsigned long long)lds;
  unsigned long long g = (unsigned long long)gaddr;
  asm volatile("global_load_async_to_lds_b128 %0, %1, off" :: "v"(l), "v"(g) : "memory");
}
__device__ __forceinline__ void wait_async0() {
  asm volatile("s_wait_asynccnt 0" ::: "memory");
}

// load a 16-element K-run of W (stride HID_) for one n, pack to bf16, 2x b128 store
__device__ __forceinline__ void load_w_column(const float* __restrict__ wp,
                                              __bf16* dst /* &Bst[n][kbase] */) {
  float f[16];
  #pragma unroll
  for (int j = 0; j < 16; ++j) f[j] = wp[(size_t)j * HID_];
  uint4v q0, q1;
  q0.x = pack2bf(f[0], f[1]);  q0.y = pack2bf(f[2], f[3]);
  q0.z = pack2bf(f[4], f[5]);  q0.w = pack2bf(f[6], f[7]);
  q1.x = pack2bf(f[8], f[9]);  q1.y = pack2bf(f[10], f[11]);
  q1.z = pack2bf(f[12], f[13]); q1.w = pack2bf(f[14], f[15]);
  *(uint4v*)dst       = q0;
  *(uint4v*)(dst + 8) = q1;
}

// ---------------------------------------------------------------------------
// Projection GEMM: Y = X(8192x1024,f32) * W(1024x1024,f32) + bias (bf16 out).
// MODE 0: split-head (B,H,S,D)   (Q, K)
// MODE 1: split-head transposed (B,H,D,S)   (V -> contiguous P*V fragments)
// Block = 128x64 tile, 128 threads (4 waves); wave = 128x16 strip = 8 WMMAs/k-step.
// Bias is folded into the accumulator init (no epilogue adds).
// ---------------------------------------------------------------------------
template <int MODE>
__global__ __launch_bounds__(128)
void gemm_proj_kernel(const float* __restrict__ X, const float* __restrict__ W,
                      const float* __restrict__ bias, __bf16* __restrict__ Y) {
  const int M0 = blockIdx.x * 128;
  const int N0 = blockIdx.y * 64;
  const int tid  = threadIdx.x;
  const int lane = tid & 31;
  const int w    = tid >> 5;
  const int half = lane >> 4;
  const int l16  = lane & 15;
  const int nn    = tid & 63;        // B-panel: column owned by this thread
  const int khalf = tid >> 6;        // 0/1 -> k 0..15 / 16..31

  __shared__ __bf16 As[128][40];     // row pitch 80B (16B aligned)
  __shared__ __bf16 Bst[64][40];     // transposed: Bst[n][k]

  const int n = N0 + w * 16 + l16;
  const float bn = bias[n];
  v8f acc[8];
  #pragma unroll
  for (int t = 0; t < 8; ++t)
    #pragma unroll
    for (int r = 0; r < 8; ++r) acc[t][r] = bn;      // bias pre-loaded into C

  for (int k0 = 0; k0 < HID_; k0 += 32) {
    __syncthreads();
    // A tile 128x32 f32 -> bf16 : 1024 float4, 8 per thread
    #pragma unroll
    for (int i = 0; i < 8; ++i) {
      int e = tid + i * 128;
      int r = e >> 3, c4 = e & 7;
      float4v f = *(const float4v*)&X[(size_t)(M0 + r) * HID_ + k0 + c4 * 4];
      uint2v p; p.x = pack2bf(f.x, f.y); p.y = pack2bf(f.z, f.w);
      *(uint2v*)&As[r][c4 * 4] = p;
    }
    // B tile 32x64 f32 -> bf16 transposed: one n-column of 16 k per thread
    load_w_column(&W[(size_t)(k0 + khalf * 16) * HID_ + N0 + nn],
                  &Bst[nn][khalf * 16]);
    __syncthreads();

    const int col = w * 16 + l16;
    const int kb  = half * 16;
    const int ab  = half * 8;
    v16bf bfrag = ld_frag(&Bst[col][kb], &Bst[col][kb + 8]);
    #pragma unroll
    for (int t = 0; t < 8; ++t) {
      const int row = t * 16 + l16;
      v16bf afrag = ld_frag(&As[row][kb], &As[row][kb + 8]);
      acc[t] = __builtin_amdgcn_wmma_f32_16x16x32_bf16(
          false, afrag, false, bfrag, (short)0, acc[t], false, false);
    }
  }

  // epilogue: constant-offset stores off one base pointer
  const int b  = M0 / S_;
  const int s0 = (M0 % S_) + half * 8;
  const int h  = N0 >> 6;
  const int d  = w * 16 + l16;
  const size_t bh = (size_t)b * HEADS_ + h;
  __bf16* yp = (MODE == 0) ? (Y + (bh * S_ + s0) * D_ + d)
                           : (Y + (bh * D_ + d) * S_ + s0);
  #pragma unroll
  for (int t = 0; t < 8; ++t)
    #pragma unroll
    for (int r = 0; r < 8; ++r) {
      if (MODE == 0) yp[(size_t)(t * 16 + r) * D_] = f2bf(acc[t][r]);
      else           yp[t * 16 + r]                = f2bf(acc[t][r]);
    }
}

// ---------------------------------------------------------------------------
// Output GEMM: out = ctx(8192x1024,bf16) * Wo(f32) + bo, fp32 output.
// A tile staged with async global->LDS copies (pure bf16 copy).
// ---------------------------------------------------------------------------
__global__ __launch_bounds__(128)
void gemm_out_kernel(const __bf16* __restrict__ X, const float* __restrict__ W,
                     const float* __restrict__ bias, float* __restrict__ Y) {
  const int M0 = blockIdx.x * 128;
  const int N0 = blockIdx.y * 64;
  const int tid  = threadIdx.x;
  const int lane = tid & 31;
  const int w    = tid >> 5;
  const int half = lane >> 4;
  const int l16  = lane & 15;
  const int nn    = tid & 63;
  const int khalf = tid >> 6;

  __shared__ __bf16 As[128][40];
  __shared__ __bf16 Bst[64][40];

  const int n = N0 + w * 16 + l16;
  const float bn = bias[n];
  v8f acc[8];
  #pragma unroll
  for (int t = 0; t < 8; ++t)
    #pragma unroll
    for (int r = 0; r < 8; ++r) acc[t][r] = bn;

  for (int k0 = 0; k0 < HID_; k0 += 32) {
    __syncthreads();
    // A tile 128x32 bf16 : 512 async b128, 4 per thread
    #pragma unroll
    for (int i = 0; i < 4; ++i) {
      int e = tid + i * 128;
      int r = e >> 2, c = e & 3;
      async_cp16(&As[r][c * 8], &X[(size_t)(M0 + r) * HID_ + k0 + c * 8]);
    }
    // B tile 32x64 f32 -> bf16 transposed: one n-column of 16 k per thread
    load_w_column(&W[(size_t)(k0 + khalf * 16) * HID_ + N0 + nn],
                  &Bst[nn][khalf * 16]);
    wait_async0();
    __syncthreads();

    const int col = w * 16 + l16;
    const int kb  = half * 16;
    const int ab  = half * 8;
    v16bf bfrag = ld_frag(&Bst[col][kb], &Bst[col][kb + 8]);
    #pragma unroll
    for (int t = 0; t < 8; ++t) {
      const int row = t * 16 + l16;
      v16bf afrag = ld_frag(&As[row][kb], &As[row][kb + 8]);
      acc[t] = __builtin_amdgcn_wmma_f32_16x16x32_bf16(
          false, afrag, false, bfrag, (short)0, acc[t], false, false);
    }
  }

  float* yp = Y + (size_t)(M0 + half * 8) * HID_ + n;
  #pragma unroll
  for (int t = 0; t < 8; ++t)
    #pragma unroll
    for (int r = 0; r < 8; ++r)
      yp[(size_t)(t * 16 + r) * HID_] = acc[t][r];
}

// ---------------------------------------------------------------------------
// Flash-attention: one block per (b, h, 64-query tile); 4 waves, 16 q-rows each.
// Q,K in (B,H,S,D); V in (B,H,D,S). Tiles staged with async global->LDS copies.
// Q fragments hoisted out of the key loop; softmax in exp2 domain.
// ---------------------------------------------------------------------------
__global__ __launch_bounds__(128)
void attn_kernel(const __bf16* __restrict__ Q, const __bf16* __restrict__ K,
                 const __bf16* __restrict__ Vt, const int* __restrict__ mask,
                 __bf16* __restrict__ ctx) {
  const int bid = blockIdx.x;
  const int qt  = bid & 31;          // S/64 tiles
  const int h   = (bid >> 5) & 15;
  const int b   = bid >> 9;
  const int tid  = threadIdx.x;
  const int lane = tid & 31;
  const int w    = tid >> 5;
  const int half = lane >> 4;
  const int l16  = lane & 15;

  const size_t bh = (size_t)b * HEADS_ + h;
  const __bf16* Qp  = Q  + (bh * S_ + (size_t)qt * 64) * D_;
  const __bf16* Kp  = K  + bh * S_ * D_;
  const __bf16* Vtp = Vt + bh * (size_t)D_ * S_;

  __shared__ __bf16 Qs[64][72];      // [q][d]   pitch 144B (16B aligned)
  __shared__ __bf16 Ks[64][72];      // [k][d]
  __shared__ __bf16 Vts[64][72];     // [d][k]
  __shared__ __bf16 Ps[4][16][72];   // per-wave P tile [q][k]

  // Q tile: 4 async b128 per thread
  #pragma unroll
  for (int i = 0; i < 4; ++i) {
    int e = tid + i * 128;
    int r = e >> 3, c = e & 7;
    async_cp16(&Qs[r][c * 8], &Qp[(size_t)r * D_ + c * 8]);
  }
  wait_async0();
  __syncthreads();

  // Q A-fragments are loop-invariant: keep them in VGPRs
  v16bf afragQ[2];
  #pragma unroll
  for (int dstep = 0; dstep < 2; ++dstep) {
    const int row = w * 16 + l16;
    const int ab  = dstep * 32 + half * 16;
    afragQ[dstep] = ld_frag(&Qs[row][ab], &Qs[row][ab + 8]);
  }

  v8f Oacc[4];
  #pragma unroll
  for (int t = 0; t < 4; ++t)
    #pragma unroll
    for (int r = 0; r < 8; ++r) Oacc[t][r] = 0.f;

  float mrun[8], lrun[8];
  #pragma unroll
  for (int r = 0; r < 8; ++r) { mrun[r] = -3.0e38f; lrun[r] = 0.f; }

  // exp2-domain softmax: fold log2(e) into the score scale (identical softmax)
  const float scale2 = 0.125f * 1.44269504088896f;   // (1/sqrt(64)) * log2(e)
  // mask base: constant-offset loads per (r,t)
  const int* mp = mask + ((size_t)b * S_ + qt * 64 + w * 16 + half * 8) * S_ + l16;
  // ctx base for epilogue
  __bf16* cp = ctx + ((size_t)b * S_ + qt * 64 + w * 16 + half * 8) * HID_ + h * D_ + l16;

  for (int kt = 0; kt < S_ / 64; ++kt) {
    __syncthreads();                 // previous iteration done reading tiles
    // K tile [k][d], V tile [d][k]: 8 async b128 per thread
    #pragma unroll
    for (int i = 0; i < 4; ++i) {
      int e = tid + i * 128;
      int r = e >> 3, c = e & 7;
      async_cp16(&Ks[r][c * 8],  &Kp[((size_t)kt * 64 + r) * D_ + c * 8]);
      async_cp16(&Vts[r][c * 8], &Vtp[(size_t)r * S_ + kt * 64 + c * 8]);
    }
    if (kt + 1 < S_ / 64)            // prefetch next mask block
      __builtin_prefetch(mp + (size_t)(kt + 1) * 64, 0, 0);
    wait_async0();
    __syncthreads();

    // S = Q(16x64) * K^T -> four 16x16 tiles, WMMA K-dim = d (2 steps of 32)
    v8f sacc[4];
    #pragma unroll
    for (int t = 0; t < 4; ++t)
      #pragma unroll
      for (int r = 0; r < 8; ++r) sacc[t][r] = 0.f;

    #pragma unroll
    for (int dstep = 0; dstep < 2; ++dstep) {
      #pragma unroll
      for (int t = 0; t < 4; ++t) {
        const int krow = t * 16 + l16;
        const int db   = dstep * 32 + half * 16;
        v16bf bfrag = ld_frag(&Ks[krow][db], &Ks[krow][db + 8]);
        sacc[t] = __builtin_amdgcn_wmma_f32_16x16x32_bf16(
            false, afragQ[dstep], false, bfrag, (short)0, sacc[t], false, false);
      }
    }

    // scale (log2 domain) + mask (constant offsets off mk)
    const int* mk = mp + kt * 64;
    #pragma unroll
    for (int t = 0; t < 4; ++t)
      #pragma unroll
      for (int r = 0; r < 8; ++r) {
        const int mv = mk[(size_t)r * S_ + t * 16];
        const float s = sacc[t][r] * scale2;
        sacc[t][r] = (mv == 0) ? -1.0e9f : s;
      }

    // online softmax (rows live in 16-lane halves -> xor masks 8..1)
    #pragma unroll
    for (int r = 0; r < 8; ++r) {
      float rm = fmaxf(fmaxf(sacc[0][r], sacc[1][r]), fmaxf(sacc[2][r], sacc[3][r]));
      #pragma unroll
      for (int off = 8; off >= 1; off >>= 1) rm = fmaxf(rm, __shfl_xor(rm, off, 32));
      const float mnew  = fmaxf(mrun[r], rm);
      const float alpha = __builtin_amdgcn_exp2f(mrun[r] - mnew);
      mrun[r] = mnew;
      float psum = 0.f;
      #pragma unroll
      for (int t = 0; t < 4; ++t) {
        const float p = __builtin_amdgcn_exp2f(sacc[t][r] - mnew);
        sacc[t][r] = p;
        psum += p;
      }
      #pragma unroll
      for (int off = 8; off >= 1; off >>= 1) psum += __shfl_xor(psum, off, 32);
      lrun[r] = lrun[r] * alpha + psum;
      #pragma unroll
      for (int t = 0; t < 4; ++t) Oacc[t][r] *= alpha;
    }

    // stage P (bf16) row-major for A fragments
    #pragma unroll
    for (int t = 0; t < 4; ++t)
      #pragma unroll
      for (int r = 0; r < 8; ++r)
        Ps[w][r + half * 8][t * 16 + l16] = f2bf(sacc[t][r]);
    __syncthreads();

    // O += P(16x64) * V(64x64), WMMA K-dim = key index (2 steps of 32)
    #pragma unroll
    for (int ks2 = 0; ks2 < 2; ++ks2) {
      const int ab = ks2 * 32 + half * 16;
      v16bf afrag = ld_frag(&Ps[w][l16][ab], &Ps[w][l16][ab + 8]);
      #pragma unroll
      for (int t = 0; t < 4; ++t) {
        const int kb  = ks2 * 32 + half * 16;
        const int col = t * 16 + l16;                  // d index
        v16bf bfrag = ld_frag(&Vts[col][kb], &Vts[col][kb + 8]);
        Oacc[t] = __builtin_amdgcn_wmma_f32_16x16x32_bf16(
            false, afrag, false, bfrag, (short)0, Oacc[t], false, false);
      }
    }
  }

  // normalize, store context in (B,S,HID) bf16 (constant offsets off cp)
  #pragma unroll
  for (int t = 0; t < 4; ++t)
    #pragma unroll
    for (int r = 0; r < 8; ++r) {
      const float v = Oacc[t][r] / lrun[r];
      cp[(size_t)r * HID_ + t * 16] = f2bf(v);
    }
}

// ---------------------------------------------------------------------------
extern "C" void kernel_launch(void* const* d_in, const int* in_sizes, int n_in,
                              void* d_out, int out_size, void* d_ws, size_t ws_size,
                              hipStream_t stream) {
  (void)in_sizes; (void)n_in; (void)out_size; (void)ws_size;
  const float* query = (const float*)d_in[0];
  const float* key   = (const float*)d_in[1];
  const float* value = (const float*)d_in[2];
  const int*   mask  = (const int*)d_in[3];
  const float* Wq = (const float*)d_in[4];
  const float* bq = (const float*)d_in[5];
  const float* Wk = (const float*)d_in[6];
  const float* bk = (const float*)d_in[7];
  const float* Wv = (const float*)d_in[8];
  const float* bv = (const float*)d_in[9];
  const float* Wo = (const float*)d_in[10];
  const float* bo = (const float*)d_in[11];
  float* out = (float*)d_out;

  const size_t nBH = (size_t)B_ * HEADS_ * S_ * D_;   // 8M elements
  __bf16* Qbh = (__bf16*)d_ws;
  __bf16* Kbh = Qbh + nBH;
  __bf16* Vbh = Kbh + nBH;   // (B,H,D,S)
  __bf16* ctx = Vbh + nBH;

  dim3 g((B_ * S_) / 128, HID_ / 64), blk(128);
  gemm_proj_kernel<0><<<g, blk, 0, stream>>>(query, Wq, bq, Qbh);
  gemm_proj_kernel<0><<<g, blk, 0, stream>>>(key,   Wk, bk, Kbh);
  gemm_proj_kernel<1><<<g, blk, 0, stream>>>(value, Wv, bv, Vbh);
  attn_kernel<<<dim3(B_ * HEADS_ * (S_ / 64)), blk, 0, stream>>>(Qbh, Kbh, Vbh, mask, ctx);
  gemm_out_kernel<<<g, blk, 0, stream>>>(ctx, Wo, bo, out);
}
